// MultiHeadAttention_33655363732092
// MI455X (gfx1250) — compile-verified, matrix operand order checked by reading, not run on hardware
//
#include <hip/hip_runtime.h>
#include <hip/hip_bf16.h>

typedef __attribute__((ext_vector_type(16))) _Float16 v16h;
typedef __attribute__((ext_vector_type(8)))  _Float16 v8h;
typedef __attribute__((ext_vector_type(8)))  float    v8f;

#define HEADS   8
#define DK      64
#define DMODEL  512
#define SEQ     4096
#define BATCH   2
#define LN_EPS  1e-5f

// ---------------------------------------------------------------------------
// WMMA helpers (wave32, v_wmma_f32_16x16x32_f16)
// A-tile 16x32 f16 per-lane layout (ISA 7.12.2): lane m=lane&15,
// kb = 8*(lane>>4); elems 0..7 = src[m][kb..kb+7], elems 8..15 = src[m][kb+16..kb+23].
// B-tile 32x16 uses the identical per-lane pattern reading B^T rows, so
// C = A * B^T with both operands row-major along K loads symmetrically.
// ---------------------------------------------------------------------------
__device__ __forceinline__ v8f wmma16(v16h a, v16h b, v8f c) {
  return __builtin_amdgcn_wmma_f32_16x16x32_f16(false, a, false, b, (short)0, c,
                                                false, false);
}

__device__ __forceinline__ v16h load_tile_f16(const _Float16* __restrict__ p0,
                                              int ld, int lane) {
  const int m  = lane & 15;
  const int kb = (lane >> 4) << 3;
  const _Float16* p = p0 + (size_t)m * ld + kb;
  v8h lo = *(const v8h*)(p);
  v8h hi = *(const v8h*)(p + 16);
  v16h r;
#pragma unroll
  for (int i = 0; i < 8; ++i) { r[i] = lo[i]; r[i + 8] = hi[i]; }
  return r;
}

__device__ __forceinline__ v16h load_tile_f32(const float* __restrict__ p0,
                                              int ld, int lane) {
  const int m  = lane & 15;
  const int kb = (lane >> 4) << 3;
  const float* p = p0 + (size_t)m * ld + kb;
  v16h r;
#pragma unroll
  for (int i = 0; i < 8; ++i) {
    r[i]     = (_Float16)p[i];
    r[i + 8] = (_Float16)p[i + 16];
  }
  return r;
}

// ---------------------------------------------------------------------------
// Projection GEMM: C = X[M,512] * W[512,512]^T   (fp32 in, f16 out)
// MODE 0: store row-major f16 [M, DMODEL]          (Q, K projections)
// MODE 1: store head-transposed f16 [B,H,DK,SEQ]   (V projection)
// One 16x16 C tile per wave; K swept in 16 steps of 32, double-buffered.
// ---------------------------------------------------------------------------
template <int MODE>
__global__ void proj_kernel(const float* __restrict__ X,
                            const float* __restrict__ W,
                            _Float16* __restrict__ Out) {
  const int lane = threadIdx.x & 31;
  const int wave = blockIdx.x * (blockDim.x >> 5) + (threadIdx.x >> 5);
  const int NT   = DMODEL / 16;          // 32 col tiles
  const int mt   = wave / NT;
  const int nt   = wave - mt * NT;
  if (mt * 16 >= BATCH * SEQ) return;

  const float* arow = X + (size_t)(mt * 16) * DMODEL;
  const float* brow = W + (size_t)(nt * 16) * DMODEL;

  v16h a_cur = load_tile_f32(arow, DMODEL, lane);
  v16h b_cur = load_tile_f32(brow, DMODEL, lane);
  v8f acc = {};
#pragma unroll 4
  for (int k = 0; k < DMODEL; k += 32) {
    const int kn = (k + 32) & (DMODEL - 1);   // wrap: branch-free tail
    v16h a_nxt = load_tile_f32(arow + kn, DMODEL, lane);
    v16h b_nxt = load_tile_f32(brow + kn, DMODEL, lane);
    acc = wmma16(a_cur, b_cur, acc);
    a_cur = a_nxt;
    b_cur = b_nxt;
  }

  const int col  = lane & 15;
  const int rsel = (lane >> 4) << 3;     // 0 or 8
  if (MODE == 0) {
    _Float16* dst =
        Out + (size_t)(mt * 16 + rsel) * DMODEL + nt * 16 + col;
#pragma unroll
    for (int r = 0; r < 8; ++r) dst[(size_t)r * DMODEL] = (_Float16)acc[r];
  } else {
    // Out[b][h][d][tok]; this lane owns 8 consecutive tokens -> one b128 store
    const int feat = nt * 16 + col;
    const int h = feat / DK, d = feat - h * DK;
    const int tokg = mt * 16 + rsel;
    const int b = tokg / SEQ, tok = tokg - b * SEQ;
    v8h pk;
#pragma unroll
    for (int r = 0; r < 8; ++r) pk[r] = (_Float16)acc[r];
    *(v8h*)(Out + (((size_t)(b * HEADS + h) * DK + d) * SEQ + tok)) = pk;
  }
}

// ---------------------------------------------------------------------------
// scores[b,h,n,m] = (1/8) * sum_d Kp[b,n,h*64+d] * Qp[b,m,h*64+d]
// Each wave: one 16x64 strip. All 2 A tiles + 8 B tiles preloaded up front
// so the 8 WMMAs drain with partial loadcnt waits.
// ---------------------------------------------------------------------------
__global__ void scores_kernel(const _Float16* __restrict__ Kp,
                              const _Float16* __restrict__ Qp,
                              float* __restrict__ attn) {
  const int lane = threadIdx.x & 31;
  long wave = (long)blockIdx.x * (blockDim.x >> 5) + (threadIdx.x >> 5);
  const int NSTRIP = SEQ / 64;           // 64
  const int MT     = SEQ / 16;           // 256
  int ns = (int)(wave % NSTRIP); wave /= NSTRIP;
  int mt = (int)(wave % MT);     wave /= MT;
  int h  = (int)(wave % HEADS);  wave /= HEADS;
  int b  = (int)wave;

  const _Float16* arow  = Kp + ((size_t)(b * SEQ + mt * 16)) * DMODEL + h * DK;
  const _Float16* brow0 = Qp + ((size_t)(b * SEQ + ns * 64)) * DMODEL + h * DK;

  v16h a0 = load_tile_f16(arow, DMODEL, lane);
  v16h a1 = load_tile_f16(arow + 32, DMODEL, lane);
  v16h bm[8];
#pragma unroll
  for (int t = 0; t < 4; ++t) {
    bm[2 * t]     = load_tile_f16(brow0 + (size_t)(t * 16) * DMODEL, DMODEL, lane);
    bm[2 * t + 1] = load_tile_f16(brow0 + (size_t)(t * 16) * DMODEL + 32, DMODEL, lane);
  }

  v8f acc[4] = {};
#pragma unroll
  for (int t = 0; t < 4; ++t) acc[t] = wmma16(a0, bm[2 * t], acc[t]);
#pragma unroll
  for (int t = 0; t < 4; ++t) acc[t] = wmma16(a1, bm[2 * t + 1], acc[t]);

  const float scale = 0.125f;            // 1/sqrt(64)
  const int col  = lane & 15;
  const int rsel = (lane >> 4) << 3;
  float* dst0 = attn +
      ((size_t)(b * HEADS + h) * SEQ + (size_t)(mt * 16 + rsel)) * SEQ +
      ns * 64 + col;
#pragma unroll
  for (int t = 0; t < 4; ++t)
#pragma unroll
    for (int r = 0; r < 8; ++r)
      dst0[(size_t)r * SEQ + t * 16] = acc[t][r] * scale;
}

// ---------------------------------------------------------------------------
// Row softmax in place over attn rows of length SEQ. 256 threads / row.
// ---------------------------------------------------------------------------
__global__ void softmax_kernel(float* __restrict__ attn) {
  float* p = attn + (size_t)blockIdx.x * SEQ;
  const int t = threadIdx.x;             // 0..255, 16 elems each
  __shared__ float red[8];

  float v[16];
  float mx = -3.402823e38f;
#pragma unroll
  for (int i = 0; i < 16; ++i) {
    v[i] = p[t + i * 256];
    mx = fmaxf(mx, v[i]);
  }
#pragma unroll
  for (int off = 16; off; off >>= 1) mx = fmaxf(mx, __shfl_xor(mx, off, 32));
  if ((t & 31) == 0) red[t >> 5] = mx;
  __syncthreads();
  mx = red[0];
#pragma unroll
  for (int i = 1; i < 8; ++i) mx = fmaxf(mx, red[i]);

  float s = 0.f;
#pragma unroll
  for (int i = 0; i < 16; ++i) {
    v[i] = __expf(v[i] - mx);
    s += v[i];
  }
#pragma unroll
  for (int off = 16; off; off >>= 1) s += __shfl_xor(s, off, 32);
  __syncthreads();
  if ((t & 31) == 0) red[t >> 5] = s;
  __syncthreads();
  s = red[0];
#pragma unroll
  for (int i = 1; i < 8; ++i) s += red[i];

  const float inv = 1.f / s;
#pragma unroll
  for (int i = 0; i < 16; ++i) p[t + i * 256] = v[i] * inv;
}

// ---------------------------------------------------------------------------
// feat[b,n,h*64+d] = sum_m attn[b,h,n,m] * Vp[b,m,h*64+d]
// attn fp32 converted to f16 on load; VpT gives contiguous B-tile rows.
// Each wave: 16 tokens x full d_k (4 accumulators). K=SEQ in 128 steps,
// software-pipelined: next step's A + 4 B tiles issued before current WMMAs.
// ---------------------------------------------------------------------------
__global__ void feat_kernel(const float* __restrict__ attn,
                            const _Float16* __restrict__ VpT,
                            _Float16* __restrict__ feat) {
  const int lane = threadIdx.x & 31;
  long wave = (long)blockIdx.x * (blockDim.x >> 5) + (threadIdx.x >> 5);
  const int MT = SEQ / 16;               // 256
  int mt = (int)(wave % MT);    wave /= MT;
  int h  = (int)(wave % HEADS); wave /= HEADS;
  int b  = (int)wave;

  const float*    arow  = attn + ((size_t)(b * HEADS + h) * SEQ + mt * 16) * SEQ;
  const _Float16* bbase = VpT + (size_t)(b * HEADS + h) * DK * SEQ;

  v16h a_cur = load_tile_f32(arow, SEQ, lane);
  v16h b_cur[4];
#pragma unroll
  for (int t = 0; t < 4; ++t)
    b_cur[t] = load_tile_f16(bbase + (size_t)(t * 16) * SEQ, SEQ, lane);

  v8f acc[4] = {};
  for (int k = 0; k < SEQ; k += 32) {
    const int kn = (k + 32) & (SEQ - 1);      // wrap: branch-free tail
    v16h a_nxt = load_tile_f32(arow + kn, SEQ, lane);
    v16h b_nxt[4];
#pragma unroll
    for (int t = 0; t < 4; ++t)
      b_nxt[t] = load_tile_f16(bbase + (size_t)(t * 16) * SEQ + kn, SEQ, lane);

#pragma unroll
    for (int t = 0; t < 4; ++t) acc[t] = wmma16(a_cur, b_cur[t], acc[t]);

    a_cur = a_nxt;
#pragma unroll
    for (int t = 0; t < 4; ++t) b_cur[t] = b_nxt[t];
  }

  const int col  = lane & 15;
  const int rsel = (lane >> 4) << 3;
  _Float16* dst = feat + (size_t)(b * SEQ + mt * 16 + rsel) * DMODEL + h * DK;
#pragma unroll
  for (int t = 0; t < 4; ++t)
#pragma unroll
    for (int r = 0; r < 8; ++r)
      dst[(size_t)r * DMODEL + t * 16 + col] = (_Float16)acc[t][r];
}

// ---------------------------------------------------------------------------
// X = feat[M,512](f16) * w_fc[512,512]^T + q   (fp32 out, residual fused)
// ---------------------------------------------------------------------------
__global__ void fc_res_kernel(const _Float16* __restrict__ A,
                              const float* __restrict__ W,
                              const float* __restrict__ q,
                              float* __restrict__ X) {
  const int lane = threadIdx.x & 31;
  const int wave = blockIdx.x * (blockDim.x >> 5) + (threadIdx.x >> 5);
  const int NT = DMODEL / 16;
  const int mt = wave / NT;
  const int nt = wave - mt * NT;
  if (mt * 16 >= BATCH * SEQ) return;

  const _Float16* arow = A + (size_t)(mt * 16) * DMODEL;
  const float*    brow = W + (size_t)(nt * 16) * DMODEL;

  v16h a_cur = load_tile_f16(arow, DMODEL, lane);
  v16h b_cur = load_tile_f32(brow, DMODEL, lane);
  v8f acc = {};
#pragma unroll 4
  for (int k = 0; k < DMODEL; k += 32) {
    const int kn = (k + 32) & (DMODEL - 1);
    v16h a_nxt = load_tile_f16(arow + kn, DMODEL, lane);
    v16h b_nxt = load_tile_f32(brow + kn, DMODEL, lane);
    acc = wmma16(a_cur, b_cur, acc);
    a_cur = a_nxt;
    b_cur = b_nxt;
  }

  const int col  = lane & 15;
  const int rsel = (lane >> 4) << 3;
  const size_t o = (size_t)(mt * 16 + rsel) * DMODEL + nt * 16 + col;
#pragma unroll
  for (int r = 0; r < 8; ++r)
    X[o + (size_t)r * DMODEL] = acc[r] + q[o + (size_t)r * DMODEL];
}

// ---------------------------------------------------------------------------
// LayerNorm over rows of 512. 128 threads/row, 4 elems/thread.
// ---------------------------------------------------------------------------
__global__ void ln_kernel(const float* __restrict__ X,
                          const float* __restrict__ gamma,
                          const float* __restrict__ beta,
                          float* __restrict__ out) {
  const float* p = X + (size_t)blockIdx.x * DMODEL;
  float* o = out + (size_t)blockIdx.x * DMODEL;
  const int t = threadIdx.x;             // 0..127
  __shared__ float red[4];

  float v[4];
  float s = 0.f;
#pragma unroll
  for (int i = 0; i < 4; ++i) { v[i] = p[t + i * 128]; s += v[i]; }
#pragma unroll
  for (int off = 16; off; off >>= 1) s += __shfl_xor(s, off, 32);
  if ((t & 31) == 0) red[t >> 5] = s;
  __syncthreads();
  s = red[0] + red[1] + red[2] + red[3];
  const float mu = s * (1.f / DMODEL);

  float ss = 0.f;
#pragma unroll
  for (int i = 0; i < 4; ++i) { float d = v[i] - mu; ss += d * d; }
#pragma unroll
  for (int off = 16; off; off >>= 1) ss += __shfl_xor(ss, off, 32);
  __syncthreads();
  if ((t & 31) == 0) red[t >> 5] = ss;
  __syncthreads();
  ss = red[0] + red[1] + red[2] + red[3];
  const float rstd = rsqrtf(ss * (1.f / DMODEL) + LN_EPS);

#pragma unroll
  for (int i = 0; i < 4; ++i) {
    const int c = t + i * 128;
    o[c] = (v[i] - mu) * rstd * gamma[c] + beta[c];
  }
}

// ---------------------------------------------------------------------------
extern "C" void kernel_launch(void* const* d_in, const int* in_sizes, int n_in,
                              void* d_out, int out_size, void* d_ws,
                              size_t ws_size, hipStream_t stream) {
  const float* q     = (const float*)d_in[0];
  const float* k     = (const float*)d_in[1];
  const float* v     = (const float*)d_in[2];
  const float* w_q   = (const float*)d_in[3];
  const float* w_k   = (const float*)d_in[4];
  const float* w_v   = (const float*)d_in[5];
  const float* w_fc  = (const float*)d_in[6];
  const float* gamma = (const float*)d_in[7];
  const float* beta  = (const float*)d_in[8];

  float* out  = (float*)d_out;                                  // [B,N,D]
  float* attn = out + (size_t)BATCH * SEQ * DMODEL;             // [B,H,N,N]

  const size_t E = (size_t)BATCH * SEQ * DMODEL;                // 4,194,304
  _Float16* Qp   = (_Float16*)d_ws;
  _Float16* Kp   = Qp + E;
  _Float16* VpT  = Kp + E;
  _Float16* feat = VpT + E;
  float*    Xres = (float*)(feat + E);

  const int projBlocks = (int)((BATCH * SEQ / 16) * (DMODEL / 16) / 4); // 4096
  proj_kernel<0><<<projBlocks, 128, 0, stream>>>(q, w_q, Qp);
  proj_kernel<0><<<projBlocks, 128, 0, stream>>>(k, w_k, Kp);
  proj_kernel<1><<<projBlocks, 128, 0, stream>>>(v, w_v, VpT);

  const int scBlocks =
      (int)((long)BATCH * HEADS * (SEQ / 16) * (SEQ / 64) / 4);  // 65536
  scores_kernel<<<scBlocks, 128, 0, stream>>>(Kp, Qp, attn);

  softmax_kernel<<<BATCH * HEADS * SEQ, 256, 0, stream>>>(attn);

  const int ftBlocks = (int)((long)BATCH * HEADS * (SEQ / 16) / 4); // 1024
  feat_kernel<<<ftBlocks, 128, 0, stream>>>(attn, VpT, feat);

  fc_res_kernel<<<projBlocks, 128, 0, stream>>>(feat, w_fc, q, Xres);

  ln_kernel<<<BATCH * SEQ, 128, 0, stream>>>(Xres, gamma, beta, out);
}
